// SimplifiedMambaBlock_14431090114820
// MI455X (gfx1250) — matrix operand order repaired and checked by reading, not compiled, not run on hardware
//
#include <hip/hip_runtime.h>
#include <hip/hip_bf16.h>

typedef __attribute__((ext_vector_type(16))) _Float16 v16h;
typedef __attribute__((ext_vector_type(4)))  _Float16 v4h;
typedef __attribute__((ext_vector_type(8)))  float    v8f;

#define BB 8
#define LL 2048
#define DD 1024
#define NN 16
#define MM (BB * LL)
#define CT 64

__device__ __forceinline__ float softplus_f(float z) {
  // numerically stable softplus
  return fmaxf(z, 0.f) + log1pf(__expf(-fabsf(z)));
}

// ---------------------------------------------------------------------------
// GEMM 1: dt = softplus(x @ Wd^T + bd)   [16384,1024] x [1024,1024]
// Block tile 128x128, 8 waves, wave tile 32x64 (2x4 WMMA 16x16), K-step 32.
// LDS holds f16 tiles already permuted into WMMA fragment order.
// ---------------------------------------------------------------------------
__global__ __launch_bounds__(256) void gemm_dt_kernel(
    const float* __restrict__ X, const float* __restrict__ Wd,
    const float* __restrict__ bd, float* __restrict__ dtout)
{
  __shared__ _Float16 sA[2][4096];   // 8 subtiles (16x32) * 512 halves
  __shared__ _Float16 sB[2][4096];   // 8 subtiles (32x16) * 512 halves

  const int tid  = threadIdx.x;
  const int w    = tid >> 5;
  const int lane = tid & 31;
  const int bm   = blockIdx.x * 128;
  const int bn   = blockIdx.y * 128;

  const v8f vzero = {0.f,0.f,0.f,0.f,0.f,0.f,0.f,0.f};
  v8f acc[2][4];
  #pragma unroll
  for (int a = 0; a < 2; ++a)
    #pragma unroll
    for (int b = 0; b < 4; ++b) acc[a][b] = vzero;

  // per-thread load/scatter mapping (4 float4 each for A and B tiles)
  int am[4], akq[4], bnn[4], aoff[4], boff[4];
  #pragma unroll
  for (int it = 0; it < 4; ++it) {
    const int f = tid + 256 * it;
    am[it]  = f >> 3;            // row within 128-row A tile
    akq[it] = (f & 7) * 4;       // k quad within 32
    bnn[it] = f >> 3;            // col within 128-col B tile
    // A fragment scatter: ISA 16-bit A 16x32 layout
    {
      const int kq = akq[it], m = am[it];
      const int mi = m >> 4, mrow = m & 15;
      const int ln = mrow | ((kq & 8) ? 16 : 0);
      const int i4 = (kq < 16) ? ((kq & 7) >> 1) : (4 + ((kq & 7) >> 1));
      aoff[it] = mi * 512 + ln * 16 + 2 * i4;
    }
    // B fragment scatter: ISA 16-bit B 32x16 layout
    {
      const int kq = akq[it], n = bnn[it];
      const int ni = n >> 4, nc = n & 15;
      const int ln = nc | ((kq & 8) ? 16 : 0);
      boff[it] = ni * 512 + ln * 16 + (kq & 7) + ((kq & 16) ? 8 : 0);
    }
  }

  float4 ar[4], br[4];
  auto g_load = [&](int k0) {
    #pragma unroll
    for (int it = 0; it < 4; ++it) {
      ar[it] = *(const float4*)(X  + (size_t)(bm + am[it])  * DD + k0 + akq[it]);
      br[it] = *(const float4*)(Wd + (size_t)(bn + bnn[it]) * DD + k0 + akq[it]);
    }
  };
  auto s_store = [&](int buf) {
    #pragma unroll
    for (int it = 0; it < 4; ++it) {
      v4h pa; pa[0] = (_Float16)ar[it].x; pa[1] = (_Float16)ar[it].y;
              pa[2] = (_Float16)ar[it].z; pa[3] = (_Float16)ar[it].w;
      *(v4h*)(&sA[buf][aoff[it]]) = pa;
      v4h pb; pb[0] = (_Float16)br[it].x; pb[1] = (_Float16)br[it].y;
              pb[2] = (_Float16)br[it].z; pb[3] = (_Float16)br[it].w;
      *(v4h*)(&sB[buf][boff[it]]) = pb;
    }
  };

  g_load(0);
  s_store(0);
  __syncthreads();

  const int asub = (w >> 1) * 2;  // first A subtile for this wave
  const int bsub = (w & 1) * 4;   // first B subtile for this wave

  for (int kt = 0; kt < 32; ++kt) {
    const int cur = kt & 1;
    if (kt + 1 < 32) g_load((kt + 1) * 32);

    v16h af[2], bf[4];
    #pragma unroll
    for (int a = 0; a < 2; ++a)
      af[a] = *(const v16h*)(&sA[cur][(asub + a) * 512 + lane * 16]);
    #pragma unroll
    for (int b = 0; b < 4; ++b)
      bf[b] = *(const v16h*)(&sB[cur][(bsub + b) * 512 + lane * 16]);

    #pragma unroll
    for (int a = 0; a < 2; ++a)
      #pragma unroll
      for (int b = 0; b < 4; ++b)
        acc[a][b] = __builtin_amdgcn_wmma_f32_16x16x32_f16(
            false, af[a], false, bf[b], (short)0, acc[a][b], false, false);

    if (kt + 1 < 32) s_store((kt + 1) & 1);
    __syncthreads();
  }

  // fused softplus epilogue; C/D layout: VGPR r -> M=r (lanes 0-15) / M=8+r
  const int mbase = bm + (w >> 1) * 32 + ((lane & 16) ? 8 : 0);
  const int nbase = bn + (w & 1) * 64 + (lane & 15);
  #pragma unroll
  for (int b = 0; b < 4; ++b) {
    const int n = nbase + b * 16;
    const float bias = bd[n];
    #pragma unroll
    for (int a = 0; a < 2; ++a) {
      #pragma unroll
      for (int r = 0; r < 8; ++r) {
        const int m = mbase + a * 16 + r;
        dtout[(size_t)m * DD + n] = softplus_f(acc[a][b][r] + bias);
      }
    }
  }
}

// ---------------------------------------------------------------------------
// GEMM 2: Bx = x @ Wb^T   [16384,1024] x [1024,16]
// Block tile 128x16, each of 8 waves does one 16x16 output tile.
// ---------------------------------------------------------------------------
__global__ __launch_bounds__(256) void gemm_bx_kernel(
    const float* __restrict__ X, const float* __restrict__ Wb,
    float* __restrict__ bxout)
{
  __shared__ _Float16 sA[2][4096];
  __shared__ _Float16 sB[2][512];

  const int tid  = threadIdx.x;
  const int w    = tid >> 5;
  const int lane = tid & 31;
  const int bm   = blockIdx.x * 128;

  v8f acc = {0.f,0.f,0.f,0.f,0.f,0.f,0.f,0.f};

  int am[4], akq[4], aoff[4];
  #pragma unroll
  for (int it = 0; it < 4; ++it) {
    const int f = tid + 256 * it;
    am[it]  = f >> 3;
    akq[it] = (f & 7) * 4;
    const int kq = akq[it], m = am[it];
    const int mi = m >> 4, mrow = m & 15;
    const int ln = mrow | ((kq & 8) ? 16 : 0);
    const int i4 = (kq < 16) ? ((kq & 7) >> 1) : (4 + ((kq & 7) >> 1));
    aoff[it] = mi * 512 + ln * 16 + 2 * i4;
  }
  const bool bact = (tid < 128);
  int bn_ = 0, bkq = 0, boff = 0;
  if (bact) {
    bn_ = tid >> 3;           // output col 0..15
    bkq = (tid & 7) * 4;      // k quad
    const int ln = bn_ | ((bkq & 8) ? 16 : 0);
    boff = ln * 16 + (bkq & 7) + ((bkq & 16) ? 8 : 0);
  }

  float4 ar[4], br;
  auto g_load = [&](int k0) {
    #pragma unroll
    for (int it = 0; it < 4; ++it)
      ar[it] = *(const float4*)(X + (size_t)(bm + am[it]) * DD + k0 + akq[it]);
    if (bact)
      br = *(const float4*)(Wb + (size_t)bn_ * DD + k0 + bkq);
  };
  auto s_store = [&](int buf) {
    #pragma unroll
    for (int it = 0; it < 4; ++it) {
      v4h pa; pa[0] = (_Float16)ar[it].x; pa[1] = (_Float16)ar[it].y;
              pa[2] = (_Float16)ar[it].z; pa[3] = (_Float16)ar[it].w;
      *(v4h*)(&sA[buf][aoff[it]]) = pa;
    }
    if (bact) {
      v4h pb; pb[0] = (_Float16)br.x; pb[1] = (_Float16)br.y;
              pb[2] = (_Float16)br.z; pb[3] = (_Float16)br.w;
      *(v4h*)(&sB[buf][boff]) = pb;
    }
  };

  g_load(0);
  s_store(0);
  __syncthreads();

  for (int kt = 0; kt < 32; ++kt) {
    const int cur = kt & 1;
    if (kt + 1 < 32) g_load((kt + 1) * 32);
    v16h af = *(const v16h*)(&sA[cur][w * 512 + lane * 16]);
    v16h bf = *(const v16h*)(&sB[cur][lane * 16]);
    acc = __builtin_amdgcn_wmma_f32_16x16x32_f16(
        false, af, false, bf, (short)0, acc, false, false);
    if (kt + 1 < 32) s_store((kt + 1) & 1);
    __syncthreads();
  }

  const int m0 = bm + w * 16 + ((lane & 16) ? 8 : 0);
  const int n  = lane & 15;
  #pragma unroll
  for (int r = 0; r < 8; ++r)
    bxout[(size_t)(m0 + r) * NN + n] = acc[r];
}

// ---------------------------------------------------------------------------
// Sequential scan over L. Block = one batch b, 16 d-channels x 16 states.
// Chunks of CT=64 timesteps staged into LDS via async global->LDS loads.
// ---------------------------------------------------------------------------
struct ScanSmem {
  float sdt[CT][16];
  float sx [CT][16];
  float sbx[CT][16];
  float sy [CT][16];
};

__global__ __launch_bounds__(256) void scan_kernel(
    const float* __restrict__ X,  const float* __restrict__ A,
    const float* __restrict__ C,  const float* __restrict__ Dp,
    const float* __restrict__ dt, const float* __restrict__ bx,
    float* __restrict__ Y)
{
  __shared__ ScanSmem sm;

  const int tid = threadIdx.x;
  const int di  = tid >> 4;          // d within block's 16 channels
  const int ni  = tid & 15;          // state index
  const int b   = blockIdx.y;
  const int d   = blockIdx.x * 16 + di;

  const float a_r  = A[d * NN + ni];
  const float c_r  = C[d * NN + ni];
  const float dp_r = Dp[d];
  float h = 0.f;

  const size_t rowbase = (size_t)b * LL;
  const int r0 = tid >> 4, c0 = tid & 15;  // staging coords (4 rows / thread)

  for (int t0 = 0; t0 < LL; t0 += CT) {
    __syncthreads();  // protect LDS reuse across chunks

    // ---- async stage dt/x/bx chunk into LDS (ASYNCcnt path) ----
    #pragma unroll
    for (int it = 0; it < 4; ++it) {
      const int r = r0 + 16 * it;
      const size_t row = rowbase + t0 + r;
      {
        const unsigned o = (unsigned)((const char*)&sm.sdt[r][c0] - (const char*)&sm);
        const float* g = dt + row * DD + (size_t)blockIdx.x * 16 + c0;
        asm volatile("global_load_async_to_lds_b32 %0, %1, off"
                     :: "v"(o), "v"(g) : "memory");
      }
      {
        const unsigned o = (unsigned)((const char*)&sm.sx[r][c0] - (const char*)&sm);
        const float* g = X + row * DD + (size_t)blockIdx.x * 16 + c0;
        asm volatile("global_load_async_to_lds_b32 %0, %1, off"
                     :: "v"(o), "v"(g) : "memory");
      }
      {
        const unsigned o = (unsigned)((const char*)&sm.sbx[r][c0] - (const char*)&sm);
        const float* g = bx + row * NN + c0;
        asm volatile("global_load_async_to_lds_b32 %0, %1, off"
                     :: "v"(o), "v"(g) : "memory");
      }
    }
    asm volatile("s_wait_asynccnt 0" ::: "memory");
    __syncthreads();

    // ---- sequential recurrence over the chunk ----
    for (int tt = 0; tt < CT; ++tt) {
      const float dtv = sm.sdt[tt][di];   // LDS broadcast across 16 lanes
      const float bxv = sm.sbx[tt][ni];
      const float e   = __expf(dtv * a_r);
      h = fmaf(e, h, dtv * bxv);          // h = exp(dt*A)*h + dt*Bx
      float p = h * c_r;
      p += __shfl_xor(p, 1);              // reduce over 16 states
      p += __shfl_xor(p, 2);              // (masks stay inside 16-lane
      p += __shfl_xor(p, 4);              //  halves of the wave32)
      p += __shfl_xor(p, 8);
      if (ni == 0)
        sm.sy[tt][di] = fmaf(dp_r, sm.sx[tt][di], p);
    }
    __syncthreads();

    // ---- coalesced writeback of y chunk ----
    #pragma unroll
    for (int it = 0; it < 4; ++it) {
      const int r = r0 + 16 * it;
      Y[(rowbase + t0 + r) * DD + (size_t)blockIdx.x * 16 + c0] = sm.sy[r][c0];
    }
  }
}

// ---------------------------------------------------------------------------
extern "C" void kernel_launch(void* const* d_in, const int* in_sizes, int n_in,
                              void* d_out, int out_size, void* d_ws, size_t ws_size,
                              hipStream_t stream) {
  (void)in_sizes; (void)n_in; (void)out_size; (void)ws_size;
  const float* x  = (const float*)d_in[0];   // [B,L,D]
  const float* A  = (const float*)d_in[1];   // [D,N]
  const float* Wb = (const float*)d_in[2];   // [N,D]
  const float* C  = (const float*)d_in[3];   // [D,N]
  const float* Dp = (const float*)d_in[4];   // [D]
  const float* Wd = (const float*)d_in[5];   // [D,D]
  const float* bd = (const float*)d_in[6];   // [D]
  float* y = (float*)d_out;                  // [B,L,D]

  float* dtws = (float*)d_ws;                       // [M,D]  64 MB
  float* bxws = dtws + (size_t)MM * DD;             // [M,N]   1 MB

  dim3 g1(MM / 128, DD / 128);
  gemm_dt_kernel<<<g1, 256, 0, stream>>>(x, Wd, bd, dtws);

  gemm_bx_kernel<<<MM / 128, 256, 0, stream>>>(x, Wb, bxws);

  dim3 g3(DD / 16, BB);
  scan_kernel<<<g3, 256, 0, stream>>>(x, A, C, Dp, dtws, bxws, y);
}
